// WindowAttention_8057358647522
// MI455X (gfx1250) — matrix operand order repaired and checked by reading, not compiled, not run on hardware
//
#include <hip/hip_runtime.h>

typedef _Float16 half_t;
typedef __attribute__((ext_vector_type(16))) _Float16 v16h;
typedef __attribute__((ext_vector_type(8)))  _Float16 v8h;
typedef __attribute__((ext_vector_type(4)))  _Float16 v4h;
typedef __attribute__((ext_vector_type(8)))  float    v8f;

#define WSZ   7
#define NH    8
#define HD    32
#define CH    256
#define N2    49
#define NTOK  64      // padded tokens per window
#define NWIN  2048
#define HW    56

static __device__ __forceinline__ v16h pack16(v8h a, v8h b) {
  v16h r;
#pragma unroll
  for (int i = 0; i < 8; ++i) { r[i] = a[i]; r[i + 8] = b[i]; }
  return r;
}

// ---------------------------------------------------------------------------
// Kernel A: QKV projection. out[m][n] = sum_k x[m][k] * qkv_w[n][k] + qkv_b[n]
// Block: one window (64 padded token rows) x 32 output columns. 256 thr, 8 waves.
// q,k written as [w][h][t(64)][d(32)] f16 ; v written transposed [w][h][d][t].
// ---------------------------------------------------------------------------
__global__ __launch_bounds__(256) void qkv_kernel(
    const float* __restrict__ x, const float* __restrict__ qkv_w,
    const float* __restrict__ qkv_b,
    half_t* __restrict__ qb, half_t* __restrict__ kb, half_t* __restrict__ vtb)
{
  __shared__ __align__(32) half_t sA[64 * 256];  // x tile (tokens x K), f16
  __shared__ __align__(32) half_t sB[32 * 256];  // weight rows (N x K), f16

  const int w     = blockIdx.x;
  const int nbase = blockIdx.y * 32;                  // 24 column groups of 32
  const int b  = w >> 6, rem = w & 63, wy = rem >> 3, wx = rem & 7;
  const int tid = threadIdx.x;

  // ---- stage x window tile (f32 -> f16), rows >= 49 zero-padded ----
  {
    int r  = tid >> 2;             // 0..63 token row
    int c0 = (tid & 3) * 64;       // 64 contiguous K per thread
    const float* src = nullptr;
    if (r < N2) {
      int ty = r / 7, tx = r - (r / 7) * 7;
      int gy = wy * 7 + ty, gx = wx * 7 + tx;
      src = x + ((size_t)(b * HW + gy) * HW + gx) * CH;
    }
#pragma unroll
    for (int i = 0; i < 16; ++i) {
      int c = c0 + i * 4;
      v4h h4 = {(_Float16)0, (_Float16)0, (_Float16)0, (_Float16)0};
      if (src) {
        float4 f = *(const float4*)(src + c);
        h4[0] = (half_t)f.x; h4[1] = (half_t)f.y; h4[2] = (half_t)f.z; h4[3] = (half_t)f.w;
      }
      *(v4h*)&sA[r * 256 + c] = h4;
    }
  }
  // ---- stage 32 weight rows (f32 -> f16) ----
  {
    int r  = tid >> 3;             // 0..31 weight row
    int c0 = (tid & 7) * 32;
    const float* wsrc = qkv_w + (size_t)(nbase + r) * CH;
#pragma unroll
    for (int i = 0; i < 8; ++i) {
      int c = c0 + i * 4;
      float4 f = *(const float4*)(wsrc + c);
      v4h h4; h4[0]=(half_t)f.x; h4[1]=(half_t)f.y; h4[2]=(half_t)f.z; h4[3]=(half_t)f.w;
      *(v4h*)&sB[r * 256 + c] = h4;
    }
  }
  __syncthreads();

  const int lane = tid & 31, wv = tid >> 5;
  const int lhi = lane >> 4, llo = lane & 15;
  const int mt = wv >> 1;        // 0..3  (row tile)
  const int nt = wv & 1;         // 0..1  (col tile)
  const int c3 = nbase >> 8;     // uniform: 0=q 1=k 2=v

  v8f acc = {};
  const int arow = mt * 16 + llo;
  const int brow = nt * 16 + llo;
#pragma unroll
  for (int k8 = 0; k8 < 8; ++k8) {
    int ka = k8 * 32 + lhi * 16;
    v8h a0 = *(v8h*)&sA[arow * 256 + ka];
    v8h a1 = *(v8h*)&sA[arow * 256 + ka + 8];
    v16h A = pack16(a0, a1);
    v16h B = *(v16h*)&sB[brow * 256 + k8 * 32 + lhi * 16];
    acc = __builtin_amdgcn_wmma_f32_16x16x32_f16(false, A, false, B,
                                                 (short)0, acc, false, false);
  }
  const int ncol = nbase + nt * 16 + llo;
  const float bias = qkv_b[ncol];
  const int h = (ncol >> 5) & 7;
  const int d = ncol & 31;
#pragma unroll
  for (int r = 0; r < 8; ++r) {
    int m = mt * 16 + r + lhi * 8;
    half_t val = (half_t)(acc[r] + bias);
    if (c3 == 0) {
      qb[(((size_t)w * NH + h) * NTOK + m) * HD + d] = val;
    } else if (c3 == 1) {
      kb[(((size_t)w * NH + h) * NTOK + m) * HD + d] = val;
    } else {
      vtb[(((size_t)w * NH + h) * HD + d) * NTOK + m] = val;
    }
  }
}

// ---------------------------------------------------------------------------
// Kernel B: windowed attention for one (window, head) per wave; 2 waves/block.
// S = q k^T * scale + relpos_bias ; masked softmax ; O = P v  -> f16 obuf.
// ---------------------------------------------------------------------------
__global__ __launch_bounds__(64) void attn_kernel(
    const half_t* __restrict__ qb, const half_t* __restrict__ kb,
    const half_t* __restrict__ vtb, const float* __restrict__ bias_table,
    half_t* __restrict__ ob)
{
  __shared__ __align__(32) float  sS[2][64 * 64];   // logits
  __shared__ __align__(32) half_t sP[2][64 * 64];   // probabilities (f16)

  const int w    = blockIdx.x;
  const int tid  = threadIdx.x;
  const int wv   = tid >> 5, lane = tid & 31;
  const int h    = blockIdx.y * 2 + wv;
  const int lhi  = lane >> 4, llo = lane & 15;
  const float scale = 0.17677669529663687f;   // 1/sqrt(32)

  const half_t* qbase = qb  + ((size_t)w * NH + h) * NTOK * HD;
  const half_t* kbase = kb  + ((size_t)w * NH + h) * NTOK * HD;
  const half_t* vbase = vtb + ((size_t)w * NH + h) * HD * NTOK;

  // preload q (A frags) and k (B frags, k stored row-major == B^T layout)
  v16h Afrag[4], Bfrag[4];
#pragma unroll
  for (int t = 0; t < 4; ++t) {
    int row = t * 16 + llo;
    v8h a0 = *(const v8h*)&qbase[row * HD + lhi * 16];
    v8h a1 = *(const v8h*)&qbase[row * HD + lhi * 16 + 8];
    Afrag[t] = pack16(a0, a1);
    Bfrag[t] = *(const v16h*)&kbase[row * HD + lhi * 16];
  }

  float* S = sS[wv];
#pragma unroll
  for (int mt = 0; mt < 4; ++mt) {
#pragma unroll
    for (int nt = 0; nt < 4; ++nt) {
      v8f c = {};
      c = __builtin_amdgcn_wmma_f32_16x16x32_f16(false, Afrag[mt], false, Bfrag[nt],
                                                 (short)0, c, false, false);
      int j  = nt * 16 + llo;
      int yj = j / 7, xj = j - yj * 7;
#pragma unroll
      for (int r = 0; r < 8; ++r) {
        int i = mt * 16 + r + lhi * 8;
        float val;
        if (j < N2 && i < N2) {
          int yi  = i / 7, xi = i - yi * 7;
          int idx = (yi - yj + 6) * 13 + (xi - xj + 6);
          val = c[r] * scale + bias_table[idx * NH + h];
        } else {
          val = -1e30f;   // mask padded keys (and dead padded query rows)
        }
        S[i * 64 + j] = val;
      }
    }
  }
  __syncthreads();

  // masked softmax: each lane owns 2 rows
  half_t* P = sP[wv];
#pragma unroll
  for (int rr = 0; rr < 2; ++rr) {
    int i = lane * 2 + rr;
    float mx = -1e30f;
    for (int j = 0; j < 64; ++j) mx = fmaxf(mx, S[i * 64 + j]);
    float sum = 0.f;
    for (int j = 0; j < 64; ++j) {
      float e = __expf(S[i * 64 + j] - mx);
      sum += e;
      P[i * 64 + j] = (half_t)e;
    }
    half_t hinv = (half_t)(1.0f / sum);
    for (int j = 0; j < 64; ++j) P[i * 64 + j] = (half_t)(P[i * 64 + j] * hinv);
  }
  __syncthreads();

  // O = P @ v  (K = 64 padded keys -> two k32 steps; v stored transposed)
#pragma unroll
  for (int mt = 0; mt < 4; ++mt) {
    int arow = mt * 16 + llo;
#pragma unroll
    for (int nt = 0; nt < 2; ++nt) {
      v8f acc = {};
#pragma unroll
      for (int kt = 0; kt < 2; ++kt) {
        int ka = kt * 32 + lhi * 16;
        v8h a0 = *(const v8h*)&P[arow * 64 + ka];
        v8h a1 = *(const v8h*)&P[arow * 64 + ka + 8];
        v16h A = pack16(a0, a1);
        v16h B = *(const v16h*)&vbase[(nt * 16 + llo) * NTOK + kt * 32 + lhi * 16];
        acc = __builtin_amdgcn_wmma_f32_16x16x32_f16(false, A, false, B,
                                                     (short)0, acc, false, false);
      }
      int d = nt * 16 + llo;
#pragma unroll
      for (int r = 0; r < 8; ++r) {
        int m = mt * 16 + r + lhi * 8;
        ob[((size_t)w * NTOK + m) * CH + h * HD + d] = (half_t)acc[r];
      }
    }
  }
}

// ---------------------------------------------------------------------------
// Kernel C: output projection + scatter back to (B, H*W, C) fp32.
// Block: one window x 64 output columns. 256 thr, 8 waves, 2 tiles each.
// ---------------------------------------------------------------------------
__global__ __launch_bounds__(256) void proj_kernel(
    const half_t* __restrict__ ob, const float* __restrict__ proj_w,
    const float* __restrict__ proj_b, float* __restrict__ out)
{
  __shared__ __align__(32) half_t sB[64 * 256];

  const int w     = blockIdx.x;
  const int nbase = blockIdx.y * 64;
  const int b = w >> 6, rem = w & 63, wy = rem >> 3, wx = rem & 7;
  const int tid = threadIdx.x;

  {
    int r  = tid >> 2;
    int c0 = (tid & 3) * 64;
    const float* wsrc = proj_w + (size_t)(nbase + r) * CH;
#pragma unroll
    for (int i = 0; i < 16; ++i) {
      int c = c0 + i * 4;
      float4 f = *(const float4*)(wsrc + c);
      v4h h4; h4[0]=(half_t)f.x; h4[1]=(half_t)f.y; h4[2]=(half_t)f.z; h4[3]=(half_t)f.w;
      *(v4h*)&sB[r * 256 + c] = h4;
    }
  }
  __syncthreads();

  const int lane = tid & 31, wv = tid >> 5;
  const int lhi = lane >> 4, llo = lane & 15;
  const half_t* abase = ob + (size_t)w * NTOK * CH;

#pragma unroll
  for (int p2 = 0; p2 < 2; ++p2) {
    int p  = wv * 2 + p2;
    int mt = p >> 2, nt = p & 3;
    int arow = mt * 16 + llo;
    v8f acc = {};
#pragma unroll
    for (int k8 = 0; k8 < 8; ++k8) {
      int ka = k8 * 32 + lhi * 16;
      v8h a0 = *(const v8h*)&abase[arow * 256 + ka];
      v8h a1 = *(const v8h*)&abase[arow * 256 + ka + 8];
      v16h A = pack16(a0, a1);
      v16h B = *(v16h*)&sB[(nt * 16 + llo) * 256 + k8 * 32 + lhi * 16];
      acc = __builtin_amdgcn_wmma_f32_16x16x32_f16(false, A, false, B,
                                                   (short)0, acc, false, false);
    }
    int ncol = nbase + nt * 16 + llo;
    float pb = proj_b[ncol];
#pragma unroll
    for (int r = 0; r < 8; ++r) {
      int m = mt * 16 + r + lhi * 8;
      if (m < N2) {
        int ty = m / 7, tx = m - (m / 7) * 7;
        int gy = wy * 7 + ty, gx = wx * 7 + tx;
        out[((size_t)(b * HW + gy) * HW + gx) * CH + ncol] = acc[r] + pb;
      }
    }
  }
}

// ---------------------------------------------------------------------------
extern "C" void kernel_launch(void* const* d_in, const int* in_sizes, int n_in,
                              void* d_out, int out_size, void* d_ws, size_t ws_size,
                              hipStream_t stream) {
  const float* x          = (const float*)d_in[0];
  // d_in[1]=H, d_in[2]=W  (fixed 56 for this problem)
  const float* qkv_w      = (const float*)d_in[3];
  const float* qkv_b      = (const float*)d_in[4];
  const float* proj_w     = (const float*)d_in[5];
  const float* proj_b     = (const float*)d_in[6];
  const float* bias_table = (const float*)d_in[7];
  float* out = (float*)d_out;

  half_t* qb  = (half_t*)d_ws;
  size_t per  = (size_t)NWIN * NH * NTOK * HD;   // 33.5M halves each
  half_t* kb  = qb + per;
  half_t* vtb = kb + per;
  half_t* ob  = vtb + per;                       // 2048*64*256 halves

  qkv_kernel<<<dim3(NWIN, 24), 256, 0, stream>>>(x, qkv_w, qkv_b, qb, kb, vtb);
  attn_kernel<<<dim3(NWIN, 4), 64, 0, stream>>>(qb, kb, vtb, bias_table, ob);
  proj_kernel<<<dim3(NWIN, 4), 256, 0, stream>>>(ob, proj_w, proj_b, out);
}
